// KANFeedForward_86397562127027
// MI455X (gfx1250) — matrix operand order repaired and checked by reading, not compiled, not run on hardware
//
#include <hip/hip_runtime.h>

#define DM    1024
#define MCH   64
#define NINNER 16

typedef __attribute__((ext_vector_type(16))) __bf16 v16bf;
typedef __attribute__((ext_vector_type(8)))  __bf16 v8bf;
typedef __attribute__((ext_vector_type(8)))  float  v8f;
typedef __attribute__((ext_vector_type(4)))  float  v4f;

// ---------------------------------------------------------------------------
// Prologue: build bf16 WMMA-friendly weight layouts in workspace.
//   WinT [64][1024]  : WinT[m][k] = Win[k][m]   (B operand for GEMM1)
//   WoutT[1024][64]  : WoutT[d][m] = Wout[m][d] (B operand for GEMM2)
// ---------------------------------------------------------------------------
__global__ __launch_bounds__(256) void prep_weights(
    const float* __restrict__ Win, const float* __restrict__ Wout,
    __bf16* __restrict__ WinT, __bf16* __restrict__ WoutT) {
  int i = blockIdx.x * 256 + threadIdx.x;     // 0 .. 65535
  int m = i >> 10, k = i & 1023;              // WinT index
  WinT[i] = (__bf16)Win[k * MCH + m];
  int d = i >> 6, m2 = i & 63;                // WoutT index
  WoutT[i] = (__bf16)Wout[m2 * DM + d];
}

// ---------------------------------------------------------------------------
// Fused KAN feed-forward. 256 threads = 8 wave32; 16 tokens per wave.
// ---------------------------------------------------------------------------
__global__ __launch_bounds__(256) void kan_fused(
    const float* __restrict__ x,
    const float* __restrict__ binp,
    const float* __restrict__ W1,
    const float* __restrict__ b1,
    const float* __restrict__ W2,
    const float* __restrict__ b2,
    const float* __restrict__ bout,
    const __bf16* __restrict__ WinT,
    const __bf16* __restrict__ WoutT,
    float* __restrict__ out)
{
  __shared__ float sW1[MCH * NINNER];
  __shared__ float sB1[MCH * NINNER];
  __shared__ float sW2[MCH * NINNER];
  __shared__ float sB2[MCH];
  __shared__ __bf16 sG[8][16 * MCH];          // per-wave g tile, row-major 16x64

  const int tid = threadIdx.x;
  for (int i = tid; i < MCH * NINNER; i += 256) {
    sW1[i] = W1[i];
    sB1[i] = b1[i];
    sW2[i] = W2[i];
  }
  if (tid < MCH) sB2[tid] = b2[tid];
  __syncthreads();

  const int wave = tid >> 5;
  const int lane = tid & 31;
  const int mcol = lane & 15;                 // N column within a 16-wide tile
  const int half = lane >> 4;                 // lane-half select for K split
  const long r0  = (long)blockIdx.x * 128 + wave * 16;

  // ---------------- GEMM1: s[16x64] = X[16x1024] @ Win[1024x64] -------------
  v8f acc[4] = {v8f{}, v8f{}, v8f{}, v8f{}};
  const float* xrow = x + (r0 + mcol) * DM;   // A: row M = lane&15
  for (int k = 0; k < DM; k += 32) {
    // A tile (16x32 bf16): lanes<16 hold K k+0..7 & k+16..23, lanes>=16 shift +8
    const float* ap = xrow + k + half * 8;
    v4f f0 = *(const v4f*)(ap + 0);
    v4f f1 = *(const v4f*)(ap + 4);
    v4f f2 = *(const v4f*)(ap + 16);
    v4f f3 = *(const v4f*)(ap + 20);
    v16bf a;
#pragma unroll
    for (int e = 0; e < 4; ++e) {
      a[e]      = (__bf16)f0[e];
      a[4 + e]  = (__bf16)f1[e];
      a[8 + e]  = (__bf16)f2[e];
      a[12 + e] = (__bf16)f3[e];
    }
#pragma unroll
    for (int t = 0; t < 4; ++t) {
      // B tile (32x16): lane holds column n = t*16+mcol; lanes<16 K 0..7 & 16..23,
      // lanes>=16 shift +8 -> in WinT row n.
      const __bf16* bp = WinT + (long)(t * 16 + mcol) * DM + k + half * 8;
      v8bf b0 = *(const v8bf*)bp;
      v8bf b1v = *(const v8bf*)(bp + 16);
      v16bf b;
#pragma unroll
      for (int e = 0; e < 8; ++e) { b[e] = b0[e]; b[8 + e] = b1v[e]; }
      acc[t] = __builtin_amdgcn_wmma_f32_16x16x32_bf16(
          false, a, false, b, (short)0, acc[t], false, false);
    }
  }

  // ---------------- per-channel 1->16->1 MLP (elementwise) ------------------
  // In the C/D layout every lane's 8 values share channel m = t*16 + mcol.
#pragma unroll
  for (int t = 0; t < 4; ++t) {
    const int ch = t * 16 + mcol;
    const float bi = binp[ch];
    v8f s = acc[t];
    v8f g;
#pragma unroll
    for (int e = 0; e < 8; ++e) { s[e] += bi; g[e] = sB2[ch]; }
    for (int j = 0; j < NINNER; ++j) {
      const float w1 = sW1[ch * NINNER + j];
      const float bb = sB1[ch * NINNER + j];
      const float w2 = sW2[ch * NINNER + j];
#pragma unroll
      for (int e = 0; e < 8; ++e) {
        float h = fmaxf(fmaf(s[e], w1, bb), 0.0f);
        g[e] = fmaf(h, w2, g[e]);
      }
    }
    // row = e + 8*half, col = ch ; row-major [16][64] bf16 in this wave's strip
    __bf16* gp = &sG[wave][0];
#pragma unroll
    for (int e = 0; e < 8; ++e) gp[(e + 8 * half) * MCH + ch] = (__bf16)g[e];
  }
  // Same-wave LDS ops are in-order; compiler inserts the dscnt wait for reuse.

  // ---------------- GEMM2: out[16x1024] = g[16x64] @ Wout[64x1024] ----------
  v16bf ga[2];
#pragma unroll
  for (int kt = 0; kt < 2; ++kt) {
    const __bf16* ap = &sG[wave][mcol * MCH + kt * 32 + half * 8];
    v8bf a0 = *(const v8bf*)ap;
    v8bf a1 = *(const v8bf*)(ap + 16);
#pragma unroll
    for (int e = 0; e < 8; ++e) { ga[kt][e] = a0[e]; ga[kt][8 + e] = a1[e]; }
  }

  for (int n0 = 0; n0 < DM; n0 += 16) {
    v8f c = v8f{};
#pragma unroll
    for (int kt = 0; kt < 2; ++kt) {
      const __bf16* bp = WoutT + (long)(n0 + mcol) * MCH + kt * 32 + half * 8;
      v8bf b0 = *(const v8bf*)bp;
      v8bf b1v = *(const v8bf*)(bp + 16);
      v16bf b;
#pragma unroll
      for (int e = 0; e < 8; ++e) { b[e] = b0[e]; b[8 + e] = b1v[e]; }
      c = __builtin_amdgcn_wmma_f32_16x16x32_bf16(
          false, ga[kt], false, b, (short)0, c, false, false);
    }
    const float bo = bout[n0 + mcol];
#pragma unroll
    for (int e = 0; e < 8; ++e) {
      out[(r0 + e + 8 * half) * DM + n0 + mcol] = c[e] + bo;
    }
  }
}

// ---------------------------------------------------------------------------
extern "C" void kernel_launch(void* const* d_in, const int* in_sizes, int n_in,
                              void* d_out, int out_size, void* d_ws, size_t ws_size,
                              hipStream_t stream) {
  const float* x    = (const float*)d_in[0];
  const float* Win  = (const float*)d_in[1];
  const float* bin_ = (const float*)d_in[2];
  const float* W1   = (const float*)d_in[3];
  const float* b1   = (const float*)d_in[4];
  const float* W2   = (const float*)d_in[5];
  const float* b2   = (const float*)d_in[6];
  const float* Wout = (const float*)d_in[7];
  const float* bout = (const float*)d_in[8];
  float* out = (float*)d_out;

  __bf16* WinT  = (__bf16*)d_ws;
  __bf16* WoutT = WinT + (size_t)DM * MCH;    // 128 KB each in workspace

  const int ntok   = in_sizes[0] / DM;        // 65536
  const int blocks = ntok / 128;              // 512 workgroups

  prep_weights<<<dim3(256), dim3(256), 0, stream>>>(Win, Wout, WinT, WoutT);
  kan_fused<<<dim3(blocks), dim3(256), 0, stream>>>(
      x, bin_, W1, b1, W2, b2, bout, WinT, WoutT, out);
}
